// HGNNPolicy_77154792505546
// MI455X (gfx1250) — compile-verified, matrix operand order checked by reading, not run on hardware
//
#include <hip/hip_runtime.h>

typedef __attribute__((ext_vector_type(16))) _Float16 v16h;
typedef __attribute__((ext_vector_type(8)))  _Float16 v8h;
typedef __attribute__((ext_vector_type(8)))  float    v8f;
typedef _Float16 half_t;

#define N_OP 200000
#define N_M  5000
#define NE   1000000
#define NA   400000
#define NP   100000
#define HID  128

// order-preserving encoding of -inf (host-side constant for amax init):
// bits(-inf)=0xFF800000, negative -> ~bits = 0x007FFFFF
#define FENC_NEG_INF 0x007FFFFFu

// ====================== utility kernels ======================

// convert W [K,N] f32 row-major -> f16 swizzled [ceil(K/32)][N][32], zero-padded.
// Swizzled so GEMM B-tile staging is a contiguous 16B-granular LDS copy and the
// per-lane WMMA fragment is two contiguous 16B chunks.
__global__ void f32_to_f16_swz_kernel(const float* __restrict__ s, half_t* __restrict__ d,
                                      int K, int N) {
  int i = blockIdx.x * blockDim.x + threadIdx.x;  // over nblk*32*N, column fastest
  int nblk = (K + 31) >> 5;
  int total = nblk * 32 * N;
  if (i >= total) return;
  int c = i % N;
  int gk = i / N;  // blk*32 + k
  int blk = gk >> 5, k = gk & 31;
  half_t v = (gk < K) ? (half_t)s[(size_t)gk * N + c] : (half_t)0.f;
  d[(size_t)blk * N * 32 + (size_t)c * 32 + k] = v;
}

__global__ void fill_u32_kernel(unsigned int* __restrict__ p, unsigned int v, int n) {
  int i = blockIdx.x * blockDim.x + threadIdx.x;
  if (i < n) p[i] = v;
}

// order-preserving float<->uint encoding for atomic float max
__device__ __forceinline__ unsigned int fenc(float f) {
  unsigned int b = __float_as_uint(f);
  return (b & 0x80000000u) ? ~b : (b | 0x80000000u);
}
__device__ __forceinline__ float fdec(unsigned int u) {
  unsigned int b = (u & 0x80000000u) ? (u & 0x7FFFFFFFu) : ~u;
  return __uint_as_float(b);
}

__device__ __forceinline__ float eluf(float v) { return v > 0.f ? v : (__expf(v) - 1.f); }

// ====================== WMMA GEMM ======================
// C[M, N] = act( (accum ? C : 0) + A[M,K](f32,lda) @ W(f16 swizzled) + bias )
// block = 128 threads (4 wave32), tile M=64 x N=NT*16, K step 32
//
// Per the ISA 16-bit 16x32 A layout, lane (half32, sub) needs K indices
// {half32*8 .. +7} then {16+half32*8 .. +7}, ascending within each chunk ->
// two contiguous 16B LDS reads (ds_load_b128). A staged row-major As[row][k];
// B staged transposed Bst[col][k] (direct copy from swizzled weights).
#define GBM 64
#define GKT 32
#define LSTR (GKT + 8)  // LDS row stride in halves (80B, multiple of 16B)

__device__ __forceinline__ v16h load_frag(const half_t* rowptr, int half32) {
  v8h lo = *(const v8h*)(rowptr + half32 * 8);
  v8h hi = *(const v8h*)(rowptr + 16 + half32 * 8);
  return __builtin_shufflevector(lo, hi, 0, 1, 2, 3, 4, 5, 6, 7,
                                 8, 9, 10, 11, 12, 13, 14, 15);
}

template <int NT>
__launch_bounds__(128)
__global__ void wmma_gemm_kernel(const float* __restrict__ A, int lda,
                                 const half_t* __restrict__ W,  // swizzled [K/32][N][32]
                                 const float* __restrict__ bias,
                                 float* __restrict__ C,
                                 int M, int K, int accum, int act) {
  constexpr int N = NT * 16;
  __shared__ __align__(16) half_t As[GBM][LSTR];   // [row][k]
  __shared__ __align__(16) half_t Bst[N][LSTR];    // [col][k] (transposed)

  const int tid = threadIdx.x;
  const int wv = tid >> 5;
  const int lane = tid & 31;
  const int m0 = blockIdx.x * GBM;
  const int half32 = lane >> 4;   // lane group (0: lanes 0-15, 1: lanes 16-31)
  const int sub = lane & 15;

  v8f acc[NT];
#pragma unroll
  for (int t = 0; t < NT; ++t)
#pragma unroll
    for (int r = 0; r < 8; ++r) acc[t][r] = 0.f;

  const bool fastA = ((lda & 3) == 0) && ((K & 31) == 0);

  for (int k0 = 0; k0 < K; k0 += GKT) {
    // ---- stage A tile (fp32 -> f16) ----
    if (fastA) {
      // vector path: 2 iters of (2x global_load_b128 + cvt + ds_store_b128)
      for (int i = tid; i < GBM * 4; i += 128) {
        int r = i >> 2, ch = (i & 3) * 8;
        int gm = m0 + r;
        v8h h;
#pragma unroll
        for (int e = 0; e < 8; ++e) h[e] = (half_t)0.f;
        if (gm < M) {
          const float* p = A + (size_t)gm * lda + k0 + ch;
          float4 f0 = *(const float4*)(p);
          float4 f1 = *(const float4*)(p + 4);
          h[0] = (half_t)f0.x; h[1] = (half_t)f0.y; h[2] = (half_t)f0.z; h[3] = (half_t)f0.w;
          h[4] = (half_t)f1.x; h[5] = (half_t)f1.y; h[6] = (half_t)f1.z; h[7] = (half_t)f1.w;
        }
        *(v8h*)&As[r][ch] = h;
      }
    } else {
      // ragged K (encoders: K=14/7): elementwise with zero pad
      for (int i = tid; i < GBM * GKT; i += 128) {
        int r = i >> 5, c = i & 31;
        int gm = m0 + r, gk = k0 + c;
        float v = (gm < M && gk < K) ? A[(size_t)gm * lda + gk] : 0.f;
        As[r][c] = (half_t)v;
      }
    }
    // ---- stage B tile: contiguous copy from swizzled weights ----
    {
      const half_t* wblk = W + (size_t)(k0 >> 5) * N * GKT;
      for (int i = tid; i < N * 4; i += 128) {
        int c = i >> 2, ch = (i & 3) * 8;
        *(v8h*)&Bst[c][ch] = *(const v8h*)(wblk + (size_t)c * GKT + ch);
      }
    }
    // CDNA5 prefetch of next A K-tile (global_prefetch_b8)
    if (k0 + GKT < K) {
      int gm = m0 + (tid >> 1);
      if (gm < M) __builtin_prefetch(&A[(size_t)gm * lda + k0 + GKT + (tid & 1) * 16], 0, 1);
    }
    __syncthreads();

    v16h afrag = load_frag(&As[wv * 16 + sub][0], half32);  // 2x ds_load_b128
    v16h bcur = load_frag(&Bst[sub][0], half32);
#pragma unroll
    for (int nt = 0; nt < NT; ++nt) {
      if (nt + 1 < NT) {
        v16h bnext = load_frag(&Bst[(nt + 1) * 16 + sub][0], half32);  // pipeline
        acc[nt] = __builtin_amdgcn_wmma_f32_16x16x32_f16(
            false, afrag, false, bcur, (short)0, acc[nt], false, false);
        bcur = bnext;
      } else {
        acc[nt] = __builtin_amdgcn_wmma_f32_16x16x32_f16(
            false, afrag, false, bcur, (short)0, acc[nt], false, false);
      }
    }
    __syncthreads();
  }

  // C/D layout: VGPR r -> M = r + half32*8, lane -> N = sub
#pragma unroll
  for (int nt = 0; nt < NT; ++nt) {
    int n = nt * 16 + sub;
#pragma unroll
    for (int r = 0; r < 8; ++r) {
      int m = m0 + wv * 16 + r + half32 * 8;
      if (m < M) {
        size_t idx = (size_t)m * N + n;
        float v = acc[nt][r];
        if (accum) v += C[idx];
        if (bias) v += bias[n];
        if (act) v = eluf(v);
        C[idx] = v;
      }
    }
  }
}

// ====================== row-dot / attention kernels ======================

// out[row] = dot(H[row, 0:128], v)
__global__ void rowdot_kernel(const float* __restrict__ H, const float* __restrict__ v,
                              float* __restrict__ out, int rows) {
  int t = blockIdx.x * blockDim.x + threadIdx.x;
  int r = t >> 5, lane = t & 31;
  if (r >= rows) return;
  const float* h = H + (size_t)r * HID;
  float s = h[lane] * v[lane] + h[lane + 32] * v[lane + 32] +
            h[lane + 64] * v[lane + 64] + h[lane + 96] * v[lane + 96];
  for (int o = 16; o; o >>= 1) s += __shfl_xor(s, o, 32);
  if (lane == 0) out[r] = s;
}

// wea[j] = dot(w_edge[j,:], att_edge)   (j = 0,1)
__global__ void wea_kernel(const float* __restrict__ w_edge, const float* __restrict__ att_edge,
                           float* __restrict__ wea) {
  int j = threadIdx.x >> 5, lane = threadIdx.x & 31;
  float s = w_edge[j * HID + lane] * att_edge[lane] +
            w_edge[j * HID + lane + 32] * att_edge[lane + 32] +
            w_edge[j * HID + lane + 64] * att_edge[lane + 64] +
            w_edge[j * HID + lane + 96] * att_edge[lane + 96];
  for (int o = 16; o; o >>= 1) s += __shfl_xor(s, o, 32);
  if (lane == 0) wea[j] = s;
}

// pass A: a_e = leaky_relu(s_src[src]+s_dst[dst]+attr.wea), atomic segment-max(dst)
__global__ void edge_a_kernel(const int* __restrict__ esrc, const int* __restrict__ edst,
                              const float* __restrict__ attr, const float* __restrict__ wea,
                              const float* __restrict__ ssrc, const float* __restrict__ sdst,
                              float* __restrict__ a_e, unsigned int* __restrict__ amax, int E) {
  int e = blockIdx.x * blockDim.x + threadIdx.x;
  if (e >= E) return;
  int s = esrc[e], d = edst[e];
  float a = ssrc[s] + sdst[d] + attr[2 * e] * wea[0] + attr[2 * e + 1] * wea[1];
  a = a > 0.f ? a : 0.2f * a;
  a_e[e] = a;
  atomicMax(&amax[d], fenc(a));
}

// decode amax in place; non-finite -> 0 (matches jnp.where(isfinite(amax), amax, 0))
__global__ void amax_fix_kernel(unsigned int* __restrict__ amax, int n) {
  int i = blockIdx.x * blockDim.x + threadIdx.x;
  if (i >= n) return;
  float f = fdec(amax[i]);
  if (!(fabsf(f) <= 3.0e38f)) f = 0.f;  // NaN or inf
  ((float*)amax)[i] = f;
}

// pass B: ex = exp(a - amax[dst]); den[dst] += ex
__global__ void edge_b_kernel(const int* __restrict__ edst, float* __restrict__ a_e,
                              const float* __restrict__ amaxf, float* __restrict__ den, int E) {
  int e = blockIdx.x * blockDim.x + threadIdx.x;
  if (e >= E) return;
  float ex = __expf(a_e[e] - amaxf[edst[e]]);
  a_e[e] = ex;
  atomicAdd(&den[edst[e]], ex);
}

// pass C: out[dst] += hs[src] * (ex / (den[dst]+1e-16)); one wave per edge, float4 slices
__global__ void edge_scatter_kernel(const int* __restrict__ esrc, const int* __restrict__ edst,
                                    const float* __restrict__ hs, const float* __restrict__ exv,
                                    const float* __restrict__ den, float* __restrict__ out, int E) {
  int t = blockIdx.x * blockDim.x + threadIdx.x;
  int e = t >> 5, lane = t & 31;
  if (e >= E) return;
  int s = esrc[e], d = edst[e];
  float alpha = exv[e] / (den[d] + 1e-16f);
  float4 v = *(const float4*)(hs + (size_t)s * HID + lane * 4);
  float* o = out + (size_t)d * HID + lane * 4;
  atomicAdd(o + 0, v.x * alpha);
  atomicAdd(o + 1, v.y * alpha);
  atomicAdd(o + 2, v.z * alpha);
  atomicAdd(o + 3, v.w * alpha);
}

// x = elu(acc + bias + x)   (GAT residual epilogue, in place on x)
__global__ void bias_res_elu_kernel(const float* __restrict__ acc, const float* __restrict__ bias,
                                    float* __restrict__ x, int n) {
  int i = blockIdx.x * blockDim.x + threadIdx.x;
  if (i >= n) return;
  x[i] = eluf(acc[i] + bias[i & (HID - 1)] + x[i]);
}

// segment-sum scatter of 128-wide rows + count (one wave per edge)
__global__ void seg_scatter_kernel(const float* __restrict__ rows, const int* __restrict__ sidx,
                                   const int* __restrict__ didx, float* __restrict__ out,
                                   float* __restrict__ cnt, int n) {
  int t = blockIdx.x * blockDim.x + threadIdx.x;
  int e = t >> 5, lane = t & 31;
  if (e >= n) return;
  int s = sidx[e], d = didx[e];
  float4 v = *(const float4*)(rows + (size_t)s * HID + lane * 4);
  float* o = out + (size_t)d * HID + lane * 4;
  atomicAdd(o + 0, v.x);
  atomicAdd(o + 1, v.y);
  atomicAdd(o + 2, v.z);
  atomicAdd(o + 3, v.w);
  if (lane == 0) atomicAdd(&cnt[d], 1.f);
}

__global__ void div_by_cnt_kernel(float* __restrict__ x, const float* __restrict__ cnt, int n) {
  int i = blockIdx.x * blockDim.x + threadIdx.x;
  if (i >= n) return;
  x[i] /= fmaxf(cnt[i >> 7], 1.f);
}

__global__ void select_by_cnt_kernel(float* __restrict__ oph, const float* __restrict__ asmh,
                                     const float* __restrict__ cnt, int n) {
  int i = blockIdx.x * blockDim.x + threadIdx.x;
  if (i >= n) return;
  if (cnt[i >> 7] > 0.f) oph[i] = asmh[i];
}

// dst[r] = src[idx[r]] (128-wide rows), out-of-range idx -> zero row (padding sentinel)
__global__ void gather_rows_kernel(const float* __restrict__ src, const int* __restrict__ idx,
                                   float* __restrict__ dst, int n, int srows) {
  int t = blockIdx.x * blockDim.x + threadIdx.x;
  int r = t >> 5, lane = t & 31;
  if (r >= n) return;
  int i = idx[r];
  float4 v = make_float4(0.f, 0.f, 0.f, 0.f);
  if ((unsigned)i < (unsigned)srows) v = *(const float4*)(src + (size_t)i * HID + lane * 4);
  *(float4*)(dst + (size_t)r * HID + lane * 4) = v;
}

// column sums (atomic) for mean embedding
__global__ void colsum_kernel(const float* __restrict__ X, int rows, float* __restrict__ out) {
  int col = threadIdx.x;  // 128
  float s = 0.f;
  for (int r = blockIdx.x; r < rows; r += gridDim.x) s += X[(size_t)r * HID + col];
  atomicAdd(&out[col], s);
}

__global__ void scale_vec_kernel(float* __restrict__ x, int n, float f) {
  int i = threadIdx.x;
  if (i < n) x[i] *= f;
}

// biastot[n] = b1[n] + sum_{k<256} ge[k] * W1[256+k, n]  (broadcast graph part of policy L1)
__global__ void pol_bias_kernel(const float* __restrict__ ge, const float* __restrict__ W1,
                                const float* __restrict__ b1, float* __restrict__ biastot) {
  int n = threadIdx.x;  // 128
  float s = b1[n];
  for (int k = 0; k < 256; ++k) s += ge[k] * W1[(size_t)(256 + k) * HID + n];
  biastot[n] = s;
}

// l1 = elu(l1 + biastot + edge_feats @ W1[512:514,:]); per-pair edge gather with sentinel
__global__ void pol_l1_finish_kernel(float* __restrict__ l1, const float* __restrict__ biastot,
                                     const float* __restrict__ W1, const float* __restrict__ attr,
                                     const int* __restrict__ peidx, int P, int E) {
  int t = blockIdx.x * blockDim.x + threadIdx.x;
  int p = t >> 5, lane = t & 31;
  if (p >= P) return;
  int ei = peidx[p];
  float e0 = 0.f, e1 = 0.f;
  if ((unsigned)ei < (unsigned)E) { e0 = attr[2 * ei]; e1 = attr[2 * ei + 1]; }
#pragma unroll
  for (int q = 0; q < 4; ++q) {
    int n = lane * 4 + q;
    size_t idx = (size_t)p * HID + n;
    float v = l1[idx] + biastot[n] + e0 * W1[(size_t)512 * HID + n] + e1 * W1[(size_t)513 * HID + n];
    l1[idx] = eluf(v);
  }
}

// logits[p] = mask ? dot(l2[p,0:64], w3) + b3 : -1e9
__global__ void logits_kernel(const float* __restrict__ l2, const float* __restrict__ w3,
                              const float* __restrict__ b3, const unsigned char* __restrict__ mask,
                              float* __restrict__ logits, int P) {
  int t = blockIdx.x * blockDim.x + threadIdx.x;
  int p = t >> 5, lane = t & 31;
  if (p >= P) return;
  const float* row = l2 + (size_t)p * 64;
  float s = row[lane] * w3[lane] + row[lane + 32] * w3[lane + 32];
  for (int o = 16; o; o >>= 1) s += __shfl_xor(s, o, 32);
  if (lane == 0) logits[p] = mask[p] ? (s + b3[0]) : -1e9f;
}

// ---- softmax reductions ----
__global__ void reduce_max_stage1(const float* __restrict__ x, int n, float* __restrict__ part) {
  __shared__ float sm[256];
  int t = threadIdx.x, i = blockIdx.x * 256 + t;
  sm[t] = (i < n) ? x[i] : -3.4e38f;
  __syncthreads();
  for (int o = 128; o; o >>= 1) { if (t < o) sm[t] = fmaxf(sm[t], sm[t + o]); __syncthreads(); }
  if (t == 0) part[blockIdx.x] = sm[0];
}

__global__ void reduce_max_final(const float* __restrict__ part, int n, float* __restrict__ g) {
  __shared__ float sm[256];
  int t = threadIdx.x;
  float v = -3.4e38f;
  for (int i = t; i < n; i += 256) v = fmaxf(v, part[i]);
  sm[t] = v;
  __syncthreads();
  for (int o = 128; o; o >>= 1) { if (t < o) sm[t] = fmaxf(sm[t], sm[t + o]); __syncthreads(); }
  if (t == 0) g[0] = sm[0];
}

__global__ void exp_stage(const float* __restrict__ logits, const float* __restrict__ gmax,
                          float* __restrict__ ex, int n, float* __restrict__ part) {
  __shared__ float sm[256];
  int t = threadIdx.x, i = blockIdx.x * 256 + t;
  float v = 0.f;
  if (i < n) { float e = __expf(logits[i] - gmax[0]); ex[i] = e; v = e; }
  sm[t] = v;
  __syncthreads();
  for (int o = 128; o; o >>= 1) { if (t < o) sm[t] += sm[t + o]; __syncthreads(); }
  if (t == 0) part[blockIdx.x] = sm[0];
}

__global__ void reduce_sum_final(const float* __restrict__ part, int n, float* __restrict__ g) {
  __shared__ float sm[256];
  int t = threadIdx.x;
  float v = 0.f;
  for (int i = t; i < n; i += 256) v += part[i];
  sm[t] = v;
  __syncthreads();
  for (int o = 128; o; o >>= 1) { if (t < o) sm[t] += sm[t + o]; __syncthreads(); }
  if (t == 0) g[0] = sm[0];
}

__global__ void normalize_kernel(const float* __restrict__ ex, const float* __restrict__ gsum,
                                 float* __restrict__ out, int n) {
  int i = blockIdx.x * blockDim.x + threadIdx.x;
  if (i < n) out[i] = ex[i] / gsum[0];
}

// value head (tiny): ge[256] -> 128 -> 64 -> 1
__global__ void value_kernel(const float* __restrict__ ge,
                             const float* __restrict__ W0, const float* __restrict__ b0,
                             const float* __restrict__ W1, const float* __restrict__ b1,
                             const float* __restrict__ W2, const float* __restrict__ b2,
                             float* __restrict__ out) {
  __shared__ float h1[128];
  __shared__ float h2[64];
  int t = threadIdx.x;  // 128
  float s = b0[t];
  for (int k = 0; k < 256; ++k) s += ge[k] * W0[(size_t)k * 128 + t];
  h1[t] = eluf(s);
  __syncthreads();
  if (t < 64) {
    float s2 = b1[t];
    for (int k = 0; k < 128; ++k) s2 += h1[k] * W1[k * 64 + t];
    h2[t] = eluf(s2);
  }
  __syncthreads();
  if (t == 0) {
    float v = b2[0];
    for (int k = 0; k < 64; ++k) v += h2[k] * W2[k];
    out[0] = v;
  }
}

// ====================== host side ======================

// params pytree flattening (jax: dict keys sorted, lists in order), base index 14:
// gat_m2o[l]: 14+7l .. (att_dst, att_edge, att_src, bias, w_dst, w_edge, w_src)
// gat_op2m[l]: 28+7l ..
// m_enc: b=42 w=43 | mlp_asm: b44 w45 b46 w47 | op_enc: b48 w49
// policy: b50 w51 b52 w53 b54 w55 | theta0: b56 w57 .. theta4: b64 w65
// value: b66 w67 b68 w69 b70 w71
enum { I_OPX = 0, I_MX, I_CEDGE, I_CATTR, I_REDGE, I_RATTR, I_ASRC, I_ADST,
       I_PREV, I_NEXT, I_POP, I_PM, I_PEIDX, I_MASK, I_PARAM0 = 14 };

struct GatP {
  const float *att_dst, *att_edge, *att_src, *bias, *w_edge;
  const half_t *w_src_h, *w_dst_h;
};

extern "C" void kernel_launch(void* const* d_in, const int* in_sizes, int n_in,
                              void* d_out, int out_size, void* d_ws, size_t ws_size,
                              hipStream_t stream) {
  (void)in_sizes; (void)n_in; (void)out_size; (void)ws_size;

  const float* op_x = (const float*)d_in[I_OPX];
  const float* machine_x = (const float*)d_in[I_MX];
  const int* cedge = (const int*)d_in[I_CEDGE];      // [2, E]: row0=op(src), row1=machine(dst)
  const float* cattr = (const float*)d_in[I_CATTR];  // [E, 2]
  const int* redge = (const int*)d_in[I_REDGE];      // [2, E]: row0=machine(src), row1=op(dst)
  const float* rattr = (const float*)d_in[I_RATTR];
  const int* asm_src = (const int*)d_in[I_ASRC];
  const int* asm_dst = (const int*)d_in[I_ADST];
  const int* prev_idx = (const int*)d_in[I_PREV];
  const int* next_idx = (const int*)d_in[I_NEXT];
  const int* pair_op = (const int*)d_in[I_POP];
  const int* pair_m = (const int*)d_in[I_PM];
  const int* pair_eidx = (const int*)d_in[I_PEIDX];
  const unsigned char* mask = (const unsigned char*)d_in[I_MASK];
  float* out = (float*)d_out;

  auto P = [&](int i) -> const float* { return (const float*)d_in[I_PARAM0 + i]; };

  // ---- workspace layout ----
  char* base = (char*)d_ws;
  size_t cur = 0;
  auto alloc = [&](size_t bytes) -> void* {
    void* p = base + cur;
    cur = (cur + bytes + 255) & ~(size_t)255;
    return p;
  };
  const size_t OPB = (size_t)N_OP * HID * sizeof(float);
  const size_t MHB = (size_t)N_M * HID * sizeof(float);

  float* op_h = (float*)alloc(OPB);
  float* machine_h = (float*)alloc(MHB);
  float* bufC = (float*)alloc(OPB);      // op-sized temp (hs / comp / gather / machine_mean / Gm)
  float* bufD = (float*)alloc(OPB);      // op-sized temp (GAT out / theta T / l1)
  float* bufE = (float*)alloc(OPB);      // op-sized temp (asm_h / theta ACC -> op_h_final)
  float* mach_tmp = (float*)alloc(MHB);  // machine-side hs/hd
  float* mach_out = (float*)alloc(MHB);  // op2m GAT accumulator
  float* sA = (float*)alloc((size_t)N_OP * 4);  // s_src
  float* sB = (float*)alloc((size_t)N_OP * 4);  // s_dst
  unsigned int* amax = (unsigned int*)alloc((size_t)N_OP * 4);
  float* den = (float*)alloc((size_t)N_OP * 4);
  float* cnt = (float*)alloc((size_t)N_OP * 4);
  float* a_e = (float*)alloc((size_t)NE * 4);
  float* logits = (float*)alloc((size_t)NP * 4);
  float* exv = (float*)alloc((size_t)NP * 4);
  float* part = (float*)alloc(4096 * 4);
  float* ge = (float*)alloc(256 * 4);
  float* biastot = (float*)alloc(128 * 4);
  float* wea = (float*)alloc(64);
  float* gmax = (float*)alloc(64);
  float* gsum = (float*)alloc(64);
  half_t* harena = (half_t*)alloc(4u << 20);

  // ---- convert + swizzle GEMM weights to f16 once ----
  size_t hcur = 0;
  auto cvt = [&](const float* src, int K, int N) -> const half_t* {
    int nblk = (K + 31) / 32;
    size_t n = (size_t)nblk * 32 * N;
    half_t* dst = harena + hcur;
    hcur = (hcur + n + 127) & ~(size_t)127;
    f32_to_f16_swz_kernel<<<(int)((n + 255) / 256), 256, 0, stream>>>(src, dst, K, N);
    return dst;
  };
  const half_t* h_wop = cvt(P(49 - 14), 14, 128);   // op_enc.w
  const half_t* h_wm = cvt(P(43 - 14), 7, 128);     // m_enc.w
  GatP g_op2m[2], g_m2o[2];
  for (int l = 0; l < 2; ++l) {
    int b = 28 - 14 + 7 * l;  // gat_op2m[l]
    g_op2m[l] = {P(b + 0), P(b + 1), P(b + 2), P(b + 3), P(b + 5),
                 cvt(P(b + 6), 128, 128), cvt(P(b + 4), 128, 128)};
    b = 14 - 14 + 7 * l;      // gat_m2o[l]
    g_m2o[l] = {P(b + 0), P(b + 1), P(b + 2), P(b + 3), P(b + 5),
                cvt(P(b + 6), 128, 128), cvt(P(b + 4), 128, 128)};
  }
  const half_t* h_wasm0 = cvt(P(45 - 14), 256, 128);
  const half_t* h_wasm1 = cvt(P(47 - 14), 128, 128);
  const half_t* h_th1 = cvt(P(59 - 14), 128, 128);
  const half_t* h_th2 = cvt(P(61 - 14), 128, 128);
  const half_t* h_th3 = cvt(P(63 - 14), 128, 128);
  const half_t* h_th4 = cvt(P(65 - 14), 128, 128);
  const half_t* h_th0 = cvt(P(57 - 14), 512, 128);
  const half_t* h_wpol1 = cvt(P(51 - 14), 256, 128);  // policy W1 rows 0:256
  const half_t* h_wpol2 = cvt(P(53 - 14), 128, 64);

  // swizzled slice offset for K-offset koff (multiple of 32): koff * N elements
  auto gemm = [&](const float* A, int lda, const half_t* W, const float* bias, float* C,
                  int M, int N, int K, int accum, int act) {
    int grid = (M + GBM - 1) / GBM;
    if (N == 128)
      wmma_gemm_kernel<8><<<grid, 128, 0, stream>>>(A, lda, W, bias, C, M, K, accum, act);
    else
      wmma_gemm_kernel<4><<<grid, 128, 0, stream>>>(A, lda, W, bias, C, M, K, accum, act);
  };
  auto wgrid = [](int rows) { return (rows * 32 + 255) / 256; };  // wave-per-row grids

  // ---- stage 0: encoders ----
  gemm(op_x, 14, h_wop, P(48 - 14), op_h, N_OP, 128, 14, 0, 0);
  gemm(machine_x, 7, h_wm, P(42 - 14), machine_h, N_M, 128, 7, 0, 0);

  // ---- stage 1: bidirectional GAT layers ----
  auto gat = [&](const float* xsrc, int Ns, float* xdst, int Nd, const int* esrc,
                 const int* edst, const float* attr, const GatP& gp, float* hs, float* hd,
                 float* outacc) {
    gemm(xsrc, 128, gp.w_src_h, nullptr, hs, Ns, 128, 128, 0, 0);
    gemm(xdst, 128, gp.w_dst_h, nullptr, hd, Nd, 128, 128, 0, 0);
    rowdot_kernel<<<wgrid(Ns), 256, 0, stream>>>(hs, gp.att_src, sA, Ns);
    rowdot_kernel<<<wgrid(Nd), 256, 0, stream>>>(hd, gp.att_dst, sB, Nd);
    wea_kernel<<<1, 64, 0, stream>>>(gp.w_edge, gp.att_edge, wea);
    fill_u32_kernel<<<(Nd + 255) / 256, 256, 0, stream>>>(amax, FENC_NEG_INF, Nd);
    edge_a_kernel<<<(NE + 255) / 256, 256, 0, stream>>>(esrc, edst, attr, wea, sA, sB, a_e, amax, NE);
    amax_fix_kernel<<<(Nd + 255) / 256, 256, 0, stream>>>(amax, Nd);
    (void)hipMemsetAsync(den, 0, (size_t)Nd * 4, stream);
    edge_b_kernel<<<(NE + 255) / 256, 256, 0, stream>>>(edst, a_e, (const float*)amax, den, NE);
    (void)hipMemsetAsync(outacc, 0, (size_t)Nd * HID * 4, stream);
    edge_scatter_kernel<<<wgrid(NE), 256, 0, stream>>>(esrc, edst, hs, a_e, den, outacc, NE);
    bias_res_elu_kernel<<<(Nd * HID + 255) / 256, 256, 0, stream>>>(outacc, gp.bias, xdst, Nd * HID);
  };
  for (int l = 0; l < 2; ++l) {
    gat(op_h, N_OP, machine_h, N_M, cedge, cedge + NE, cattr, g_op2m[l], bufC, mach_tmp, mach_out);
    gat(machine_h, N_M, op_h, N_OP, redge, redge + NE, rattr, g_m2o[l], mach_tmp, bufC, bufD);
  }

  // ---- stage 2: assembly MLP via segment mean ----
  (void)hipMemsetAsync(bufC, 0, OPB, stream);
  (void)hipMemsetAsync(cnt, 0, (size_t)N_OP * 4, stream);
  seg_scatter_kernel<<<wgrid(NA), 256, 0, stream>>>(op_h, asm_src, asm_dst, bufC, cnt, NA);
  div_by_cnt_kernel<<<(N_OP * HID + 255) / 256, 256, 0, stream>>>(bufC, cnt, N_OP * HID);
  gemm(op_h, 128, h_wasm0, nullptr, bufD, N_OP, 128, 128, 0, 0);                // concat part 1
  gemm(bufC, 128, h_wasm0 + 128 * 128, P(44 - 14), bufD, N_OP, 128, 128, 1, 1); // part 2 + elu
  gemm(bufD, 128, h_wasm1, P(46 - 14), bufE, N_OP, 128, 128, 0, 0);
  select_by_cnt_kernel<<<(N_OP * HID + 255) / 256, 256, 0, stream>>>(op_h, bufE, cnt, N_OP * HID);

  // ---- stage 3: theta MLP; accumulate theta0 over K-slices (no h4 materialization) ----
  gather_rows_kernel<<<wgrid(N_OP), 256, 0, stream>>>(op_h, prev_idx, bufC, N_OP, N_OP);
  gemm(bufC, 128, h_th1, P(58 - 14), bufD, N_OP, 128, 128, 0, 1);
  gemm(bufD, 128, h_th0, nullptr, bufE, N_OP, 128, 128, 0, 0);
  gather_rows_kernel<<<wgrid(N_OP), 256, 0, stream>>>(op_h, next_idx, bufC, N_OP, N_OP);
  gemm(bufC, 128, h_th2, P(60 - 14), bufD, N_OP, 128, 128, 0, 1);
  gemm(bufD, 128, h_th0 + 128 * 128, nullptr, bufE, N_OP, 128, 128, 1, 0);
  (void)hipMemsetAsync(bufC, 0, OPB, stream);
  (void)hipMemsetAsync(cnt, 0, (size_t)N_OP * 4, stream);
  seg_scatter_kernel<<<wgrid(NE), 256, 0, stream>>>(machine_h, cedge + NE, cedge, bufC, cnt, NE);
  div_by_cnt_kernel<<<(N_OP * HID + 255) / 256, 256, 0, stream>>>(bufC, cnt, N_OP * HID);
  gemm(bufC, 128, h_th3, P(62 - 14), bufD, N_OP, 128, 128, 0, 1);
  gemm(bufD, 128, h_th0 + 256 * 128, nullptr, bufE, N_OP, 128, 128, 1, 0);
  gemm(op_h, 128, h_th4, P(64 - 14), bufD, N_OP, 128, 128, 0, 1);
  gemm(bufD, 128, h_th0 + 384 * 128, P(56 - 14), bufE, N_OP, 128, 128, 1, 1);  // + bias, elu
  float* op_f = bufE;  // final op embeddings

  // ---- graph embedding (column means) ----
  (void)hipMemsetAsync(ge, 0, 256 * 4, stream);
  colsum_kernel<<<256, 128, 0, stream>>>(op_f, N_OP, ge);
  colsum_kernel<<<64, 128, 0, stream>>>(machine_h, N_M, ge + 128);
  scale_vec_kernel<<<1, 128, 0, stream>>>(ge, 128, 1.f / (float)N_OP);
  scale_vec_kernel<<<1, 128, 0, stream>>>(ge + 128, 128, 1.f / (float)N_M);

  // ---- policy head ----
  pol_bias_kernel<<<1, 128, 0, stream>>>(ge, P(51 - 14), P(50 - 14), biastot);
  gather_rows_kernel<<<wgrid(NP), 256, 0, stream>>>(op_f, pair_op, op_h, NP, N_OP);       // Gop
  gather_rows_kernel<<<wgrid(NP), 256, 0, stream>>>(machine_h, pair_m, bufC, NP, N_M);    // Gm
  gemm(op_h, 128, h_wpol1, nullptr, bufD, NP, 128, 128, 0, 0);
  gemm(bufC, 128, h_wpol1 + 128 * 128, nullptr, bufD, NP, 128, 128, 1, 0);
  pol_l1_finish_kernel<<<wgrid(NP), 256, 0, stream>>>(bufD, biastot, P(51 - 14), cattr,
                                                      pair_eidx, NP, NE);
  gemm(bufD, 128, h_wpol2, P(52 - 14), op_h, NP, 64, 128, 0, 1);  // l2 (P x 64) in op_h buf
  logits_kernel<<<wgrid(NP), 256, 0, stream>>>(op_h, P(55 - 14), P(54 - 14), mask, logits, NP);

  // softmax over all P pairs -> out[0:P]
  int nb = (NP + 255) / 256;
  reduce_max_stage1<<<nb, 256, 0, stream>>>(logits, NP, part);
  reduce_max_final<<<1, 256, 0, stream>>>(part, nb, gmax);
  exp_stage<<<nb, 256, 0, stream>>>(logits, gmax, exv, NP, part);
  reduce_sum_final<<<1, 256, 0, stream>>>(part, nb, gsum);
  normalize_kernel<<<nb, 256, 0, stream>>>(exv, gsum, out, NP);

  // ---- value head -> out[P] ----
  value_kernel<<<1, 128, 0, stream>>>(ge, P(67 - 14), P(66 - 14), P(69 - 14), P(68 - 14),
                                      P(71 - 14), P(70 - 14), out + NP);
}
